// AttentionCrossLayer_14783277433008
// MI455X (gfx1250) — matrix operand order repaired and checked by reading, not compiled, not run on hardware
//
#include <hip/hip_runtime.h>

// MI455X / gfx1250, wave32. WMMA 16x16x32 bf16 with f32 accumulate.
typedef __attribute__((ext_vector_type(16))) __bf16 v16bf;
typedef __attribute__((ext_vector_type(8)))  float  v8f;

#define L_LAYERS 3
#define BROWS    32768
#define DDIM     1024
#define HKDIM    512

__device__ __forceinline__ unsigned bf16_hi_bits(float f) {
    union { float f; unsigned u; } c; c.f = f;
    return c.u >> 16;                      // truncate to bf16 bit pattern
}
__device__ __forceinline__ float bits_to_f32(unsigned hi16) {
    union { unsigned u; float f; } c; c.u = hi16 << 16;
    return c.f;
}

// ---------------------------------------------------------------------------
// Kernel 1: tiny precompute (single workgroup).
//   t_i  = Wo_i (512xD) @ cw_i            (3 x 512)
//   u_i  = Wv_i (Dx512) @ t_i             (3 x 1024)
//   c_i  = bv_i . t_i + bo_i . cw_i       (3 scalars -> csum)
//   cbsum= sum_i cb_i                     (1024)
//   bpack: u_{hi,lo} packed per lane in this file's contiguous K order (K = 16*half + s) for the
//          WMMA bf16 B operand (32 K-chunks x 32 lanes x 8 dwords).
//          Columns 0..2 = bf16_hi(u_i), columns 3..5 = bf16_lo(u_i), rest 0.
// ---------------------------------------------------------------------------
__global__ __launch_bounds__(1024) void athena_precompute(
    const float* __restrict__ wv, const float* __restrict__ bv,
    const float* __restrict__ wo, const float* __restrict__ bo,
    const float* __restrict__ cw, const float* __restrict__ cb,
    unsigned* __restrict__ bpack, float* __restrict__ cbsum,
    float* __restrict__ csum)
{
    __shared__ float t[L_LAYERS][HKDIM];
    __shared__ float u[L_LAYERS][DDIM];
    __shared__ float cpart[L_LAYERS];
    const int tid = threadIdx.x;

    // t_i[j] = sum_d wo[i][j][d] * cw[i][d]
    for (int idx = tid; idx < L_LAYERS * HKDIM; idx += 1024) {
        const int i = idx / HKDIM, j = idx % HKDIM;
        const float* wor = wo + (size_t)(i * HKDIM + j) * DDIM;
        const float* cwi = cw + i * DDIM;
        float s = 0.f;
        for (int d = 0; d < DDIM; ++d) s = fmaf(wor[d], cwi[d], s);
        t[i][j] = s;
    }
    __syncthreads();

    // u_i[d] = sum_j wv[i][d][j] * t[i][j]
    for (int idx = tid; idx < L_LAYERS * DDIM; idx += 1024) {
        const int i = idx / DDIM, k = idx % DDIM;
        const float* wvr = wv + (size_t)(i * DDIM + k) * HKDIM;
        float s = 0.f;
        for (int j = 0; j < HKDIM; ++j) s = fmaf(wvr[j], t[i][j], s);
        u[i][k] = s;
    }

    if (tid < L_LAYERS) {
        float s = 0.f;
        for (int j = 0; j < HKDIM; ++j) s = fmaf(bv[tid * HKDIM + j], t[tid][j], s);
        for (int d = 0; d < DDIM;  ++d) s = fmaf(bo[tid * DDIM + d], cw[tid * DDIM + d], s);
        cpart[tid] = s;
    }
    for (int d = tid; d < DDIM; d += 1024)
        cbsum[d] = cb[d] + cb[DDIM + d] + cb[2 * DDIM + d];
    __syncthreads();
    if (tid == 0) csum[0] = cpart[0] + cpart[1] + cpart[2];

    // Pack B operand: lane half -> K group {0..15}/{16..31}; dword j holds
    // K = 16*half + 2j (low 16b) and +1 (high 16b); lane%16 = column N.
    for (int idx = tid; idx < 32 * 32 * 8; idx += 1024) {
        const int kb   = idx >> 8;
        const int lane = (idx >> 3) & 31;
        const int j    = idx & 7;
        const int half = lane >> 4, n = lane & 15;
        const int k0   = kb * 32 + 16 * half + 2 * j;
        unsigned dw = 0;
        #pragma unroll
        for (int e = 0; e < 2; ++e) {
            const int k = k0 + e;
            unsigned b = 0;
            if (n < 3) {
                b = bf16_hi_bits(u[n][k]);                       // u_hi
            } else if (n < 6) {
                const float uv = u[n - 3][k];
                const float hi = bits_to_f32(bf16_hi_bits(uv));
                b = bf16_hi_bits(uv - hi);                       // u_lo
            }
            dw |= b << (16 * e);
        }
        bpack[idx] = dw;
    }
}

// ---------------------------------------------------------------------------
// Kernel 2: streaming pass. One 16-row tile per wave (8 waves/block).
//   Phase 1: D[m,n] += sum_K x_{hi,lo}[m,K] * Bpack[K,n]  via v_wmma bf16
//            -> xw factors per row (cols 0..5 summed), near-f32 accuracy.
//   Phase 2: out[b,d] = x[b,d] * (1 + sum_i xw_i[b]) + cbsum[d], coalesced.
// ---------------------------------------------------------------------------
__global__ __launch_bounds__(256) void athena_main(
    const float* __restrict__ x, const unsigned* __restrict__ bpack,
    const float* __restrict__ cbsum, const float* __restrict__ csum,
    float* __restrict__ out)
{
    __shared__ unsigned sB[32 * 32 * 8];   // 32 KB packed B operand table
    __shared__ float sD[8][8][32];         // per-wave WMMA D staging
    __shared__ float sXW[8][16];           // per-row multiplicative factors
    __shared__ float sCB[DDIM];

    const int tid = threadIdx.x;
    for (int idx = tid; idx < 32 * 32 * 8; idx += 256) sB[idx] = bpack[idx];
    for (int d = tid; d < DDIM; d += 256) sCB[d] = cbsum[d];
    __syncthreads();

    const int wave = tid >> 5;
    const int lane = tid & 31;
    const int m    = lane & 15;            // A-operand row for this lane
    const int half = lane >> 4;            // lane half selects K sub-group
    const size_t rowbase = ((size_t)blockIdx.x * 8 + wave) * 16;
    const float* xrow = x + (rowbase + m) * DDIM;

    v8f acc = {0.f, 0.f, 0.f, 0.f, 0.f, 0.f, 0.f, 0.f};
    for (int kb = 0; kb < 32; ++kb) {
        const int k0 = kb * 32 + 16 * half; // VGPRs0-7: K=16*half+0..15
        float av[16];
        const float4 q0 = *(const float4*)(xrow + k0);
        const float4 q1 = *(const float4*)(xrow + k0 + 4);
        const float4 q2 = *(const float4*)(xrow + k0 + 8);
        const float4 q3 = *(const float4*)(xrow + k0 + 12);
        av[0]=q0.x; av[1]=q0.y; av[2]=q0.z;  av[3]=q0.w;
        av[4]=q1.x; av[5]=q1.y; av[6]=q1.z;  av[7]=q1.w;
        av[8]=q2.x; av[9]=q2.y; av[10]=q2.z; av[11]=q2.w;
        av[12]=q3.x; av[13]=q3.y; av[14]=q3.z; av[15]=q3.w;

        union { v16bf v; unsigned u[8]; } ahi, alo;
        #pragma unroll
        for (int e = 0; e < 8; ++e) {
            const unsigned h0 = bf16_hi_bits(av[2 * e]);
            const unsigned h1 = bf16_hi_bits(av[2 * e + 1]);
            const unsigned l0 = bf16_hi_bits(av[2 * e]     - bits_to_f32(h0));
            const unsigned l1 = bf16_hi_bits(av[2 * e + 1] - bits_to_f32(h1));
            ahi.u[e] = h0 | (h1 << 16);
            alo.u[e] = l0 | (l1 << 16);
        }
        union { v16bf v; uint4 q[2]; } bb;
        const uint4* bp = (const uint4*)&sB[kb * 256 + lane * 8];
        bb.q[0] = bp[0];
        bb.q[1] = bp[1];

        // (neg_a, A, neg_b, B, c_mod, C, reuse_a, reuse_b)
        acc = __builtin_amdgcn_wmma_f32_16x16x32_bf16(
            false, ahi.v, false, bb.v, (short)0, acc, false, false);
        acc = __builtin_amdgcn_wmma_f32_16x16x32_bf16(
            false, alo.v, false, bb.v, (short)0, acc, false, false);
    }

    #pragma unroll
    for (int r = 0; r < 8; ++r) sD[wave][r][lane] = acc[r];
    __syncthreads();

    if (lane < 16) {
        // D layout: row m -> VGPR m%8, lanes 16*(m/8)+n
        float s = 0.f;
        #pragma unroll
        for (int n = 0; n < 6; ++n) s += sD[wave][lane & 7][(lane >> 3) * 16 + n];
        sXW[wave][lane] = s + csum[0];     // sum_i (x.u_i) + sum_i c_i
    }
    __syncthreads();

    const float* xr2  = x   + rowbase * DDIM;
    float*       outr = out + rowbase * DDIM;
    for (int mm = 0; mm < 16; ++mm) {
        const float factor = 1.0f + sXW[wave][mm];
        for (int d = lane * 4; d < DDIM; d += 128) {
            const float4 xv = *(const float4*)(xr2 + (size_t)mm * DDIM + d);
            const float4 cv = *(const float4*)&sCB[d];
            float4 o;
            o.x = fmaf(xv.x, factor, cv.x);
            o.y = fmaf(xv.y, factor, cv.y);
            o.z = fmaf(xv.z, factor, cv.z);
            o.w = fmaf(xv.w, factor, cv.w);
            *(float4*)(outr + (size_t)mm * DDIM + d) = o;
        }
    }
}

extern "C" void kernel_launch(void* const* d_in, const int* in_sizes, int n_in,
                              void* d_out, int out_size, void* d_ws, size_t ws_size,
                              hipStream_t stream) {
    (void)in_sizes; (void)n_in; (void)out_size; (void)ws_size;
    const float* x  = (const float*)d_in[0];
    // d_in[1..4] = wq,bq,wk,bk : dead (softmax over length-1 axis == 1)
    const float* wv = (const float*)d_in[5];
    const float* bv = (const float*)d_in[6];
    const float* wo = (const float*)d_in[7];
    const float* bo = (const float*)d_in[8];
    const float* cw = (const float*)d_in[9];
    const float* cb = (const float*)d_in[10];

    unsigned* bpack = (unsigned*)d_ws;                              // 32 KB
    float*    cbsum = (float*)((char*)d_ws + 32 * 32 * 8 * 4);      //  4 KB
    float*    csum  = (float*)((char*)d_ws + 32 * 32 * 8 * 4 + DDIM * 4);

    athena_precompute<<<1, 1024, 0, stream>>>(wv, bv, wo, bo, cw, cb,
                                              bpack, cbsum, csum);

    const int tiles  = BROWS / 16;         // 2048 tiles of 16 rows
    const int blocks = tiles / 8;          // 8 waves per block
    athena_main<<<blocks, 256, 0, stream>>>(x, bpack, cbsum, csum, (float*)d_out);
}
